// EfficientSelfAttention_86363202388622
// MI455X (gfx1250) — compile-verified, matrix operand order checked by reading, not run on hardware
//
#include <hip/hip_runtime.h>
#include <hip/hip_bf16.h>

// Problem constants (from reference): B=2, C=256, CK=32, N=T*H*W=65536
#define NB    2
#define CCH   256
#define CKCH  32
#define NSP   65536
#define CHUNK 512    // spatial positions reduced per block in k_sz
#define ESTRIDE 520  // padded LDS row stride (bf16) -> 4-dword bank skew/row

typedef __attribute__((ext_vector_type(16))) __bf16 v16bf;
typedef __attribute__((ext_vector_type(8)))  float  v8f;

// Native conversion: lowers to v_cvt_pk_bf16_f32 on gfx1250 (asm-verified)
__device__ __forceinline__ __bf16 to_bf16(float f) { return (__bf16)f; }

// ds_swizzle_b32, group-of-32 xor mode: offset = {xor[14:10], or[9:5]=0,
// and[4:0]=0x1F}. One DS op per lane exchange; replaces generic __shfl_xor's
// ds_bpermute + lane-mask VALU chain.
template <int XORM>
__device__ __forceinline__ float swz_xor(float v) {
  int i = __builtin_amdgcn_ds_swizzle(__builtin_bit_cast(int, v),
                                      (XORM << 10) | 0x1F);
  return __builtin_bit_cast(float, i);
}

// ISA 7.12.2: 16-bit A/B fragment, lane L holds m/n = L%16; element e maps to
// k = (e<8 ? e : e+8) + 8*(L>=16)
__device__ __forceinline__ int k_of(int e, int half) {
  return ((e < 8) ? e : (e + 8)) + half * 8;
}

__device__ __forceinline__ v8f wmma_bf16(v16bf a, v16bf b, v8f c) {
  return __builtin_amdgcn_wmma_f32_16x16x32_bf16(false, a, false, b, (short)0, c,
                                                 false, false);
}

// ------------------------------------------------- init scratch + weight cvt
__global__ void k_init(float* __restrict__ S, float* __restrict__ Z,
                       float* __restrict__ kmax,
                       const float* __restrict__ Wk, const float* __restrict__ Wq,
                       __bf16* __restrict__ Wkbf, __bf16* __restrict__ Wqbf) {
  int i = blockIdx.x * blockDim.x + threadIdx.x;
  if (i < NB * CKCH * CCH) S[i] = 0.0f;
  if (i < NB * CKCH) { Z[i] = 0.0f; kmax[i] = -3.0e38f; }
  if (i < CKCH * CCH) {               // one-time bf16 copies of Wk / Wq (16 KB each)
    Wkbf[i] = to_bf16(Wk[i]);
    Wqbf[i] = to_bf16(Wq[i]);
  }
}

// ------------------------------------------- pass 1: klogit, kmax, q-softmax
// 8 waves/block; each wave owns one 16-column spatial tile.
// WMMA: [32x256] Wk|Wq (bf16, pre-converted) x [256x16] x-tile.
// kmax: wave swizzle-reduce -> LDS atomicMax -> one global atomicMax per row.
__global__ __launch_bounds__(256) void k_qk(
    const float* __restrict__ x,  const __bf16* __restrict__ Wqbf,
    const float* __restrict__ bq, const __bf16* __restrict__ Wkbf,
    const float* __restrict__ bk, float* __restrict__ klogit,
    __bf16* __restrict__ qsmT, float* __restrict__ kmax)
{
  __shared__ float kmaxl[CKCH];

  const int tid  = threadIdx.x;
  const int lane = tid & 31;
  const int wave = tid >> 5;
  const int half = lane >> 4;
  const int col  = lane & 15;
  const int b    = blockIdx.y;
  const int n0   = (blockIdx.x * 8 + wave) * 16;
  const float* xb = x + (size_t)b * CCH * NSP;

  if (tid < CKCH) kmaxl[tid] = -3.0e38f;
  __syncthreads();

  v8f accK0 = {}, accK1 = {}, accQ0 = {}, accQ1 = {};

  for (int kc = 0; kc < CCH; kc += 32) {
    v16bf bf;
    #pragma unroll
    for (int e = 0; e < 16; ++e) {
      int k = kc + k_of(e, half);
      bf[e] = to_bf16(xb[(size_t)k * NSP + n0 + col]);  // coalesced across lanes
    }
    v16bf aK0, aK1, aQ0, aQ1;
    #pragma unroll
    for (int e = 0; e < 16; ++e) {
      int k = kc + k_of(e, half);            // contiguous bf16 runs per lane
      aK0[e] = Wkbf[(size_t)(col     ) * CCH + k];
      aK1[e] = Wkbf[(size_t)(col + 16) * CCH + k];
      aQ0[e] = Wqbf[(size_t)(col     ) * CCH + k];
      aQ1[e] = Wqbf[(size_t)(col + 16) * CCH + k];
    }
    accK0 = wmma_bf16(aK0, bf, accK0);
    accK1 = wmma_bf16(aK1, bf, accK1);
    accQ0 = wmma_bf16(aQ0, bf, accQ0);
    accQ1 = wmma_bf16(aQ1, bf, accQ1);
  }

  // key logits: + bias, store fp32, fold row-max into LDS
  #pragma unroll
  for (int mt = 0; mt < 2; ++mt) {
    v8f acc = mt ? accK1 : accK0;
    #pragma unroll
    for (int r = 0; r < 8; ++r) {
      int d = mt * 16 + half * 8 + r;
      float kl = acc[r] + bk[d];
      klogit[((size_t)(b * CKCH + d)) * NSP + n0 + col] = kl;
      float m = kl;                                   // max over 16 columns
      m = fmaxf(m, swz_xor<1>(m));
      m = fmaxf(m, swz_xor<2>(m));
      m = fmaxf(m, swz_xor<4>(m));
      m = fmaxf(m, swz_xor<8>(m));
      if (col == 0) atomicMax(&kmaxl[d], m);          // LDS: ds_max_num_f32
    }
  }

  // query: softmax over the 32 channels (split between lane L and L^16)
  float q[16];
  float qmax = -3.0e38f;
  #pragma unroll
  for (int mt = 0; mt < 2; ++mt) {
    v8f acc = mt ? accQ1 : accQ0;
    #pragma unroll
    for (int r = 0; r < 8; ++r) {
      int d = mt * 16 + half * 8 + r;
      float ql = acc[r] + bq[d];
      q[mt * 8 + r] = ql;
      qmax = fmaxf(qmax, ql);
    }
  }
  qmax = fmaxf(qmax, swz_xor<16>(qmax));
  float qsum = 0.0f;
  #pragma unroll
  for (int i = 0; i < 16; ++i) { q[i] = __expf(q[i] - qmax); qsum += q[i]; }
  qsum += swz_xor<16>(qsum);
  float inv = 1.0f / qsum;
  size_t qb = ((size_t)b * NSP + n0 + col) * CKCH;   // transposed: [b][n][32]
  #pragma unroll
  for (int mt = 0; mt < 2; ++mt)
    #pragma unroll
    for (int r = 0; r < 8; ++r)
      qsmT[qb + (size_t)(mt * 16 + half * 8 + r)] = to_bf16(q[mt * 8 + r] * inv);

  // one global atomicMax per row per block
  __syncthreads();
  if (tid < CKCH) atomicMax(&kmax[b * CKCH + tid], kmaxl[tid]);
}

// ------------------- pass 2: S[d,c] = sum_n exp(kl-m)*x[c,n];  Z[d] = sum exp
// Phase 1: block computes E = exp(klogit - m) ONCE into LDS (bf16, padded
// stride) and accumulates Z. Phase 2: 8 waves each own 32 c-channels, read
// A-fragments from LDS (ds_load), B-fragments (x) from global.
__global__ __launch_bounds__(256) void k_sz(
    const float* __restrict__ x, const float* __restrict__ klogit,
    const float* __restrict__ kmax, float* __restrict__ S, float* __restrict__ Z)
{
  __shared__ __bf16 Elds[CKCH * ESTRIDE];          // 32*520*2 = 33280 B

  const int tid = threadIdx.x;
  const int b   = blockIdx.y;
  const int nb0 = blockIdx.x * CHUNK;
  const float* xb = x + (size_t)b * CCH * NSP;

  // ---- phase 1: E into LDS + Z ----
  {
    int d   = tid & 31;
    int seg = tid >> 5;                            // 8 segments of 64 columns
    const float* klrow =
        klogit + ((size_t)(b * CKCH + d)) * NSP + nb0 + seg * 64;
    float m = kmax[b * CKCH + d];
    float z = 0.0f;
    #pragma unroll 4
    for (int j = 0; j < 64; ++j) {
      float ev = __expf(klrow[j] - m);
      z += ev;
      Elds[d * ESTRIDE + seg * 64 + j] = to_bf16(ev);
    }
    atomicAdd(&Z[b * CKCH + d], z);
  }
  __syncthreads();

  // ---- phase 2: WMMA over the chunk ----
  const int lane = tid & 31;
  const int wave = tid >> 5;
  const int half = lane >> 4;
  const int col  = lane & 15;
  const int c0   = wave * 32;

  v8f a00 = {}, a01 = {}, a10 = {}, a11 = {};

  for (int kk = 0; kk < CHUNK; kk += 32) {
    __builtin_prefetch(&xb[(size_t)(c0 + col) * NSP + nb0 + kk + 64], 0, 1);
    v16bf fa0, fa1, fb0, fb1;
    #pragma unroll
    for (int e = 0; e < 16; ++e) {
      int k = kk + k_of(e, half);
      fa0[e] = Elds[(col     ) * ESTRIDE + k];     // LDS, 16B runs per lane
      fa1[e] = Elds[(col + 16) * ESTRIDE + k];
      fb0[e] = to_bf16(xb[(size_t)(c0 + col     ) * NSP + nb0 + k]);
      fb1[e] = to_bf16(xb[(size_t)(c0 + 16 + col) * NSP + nb0 + k]);
    }
    a00 = wmma_bf16(fa0, fb0, a00);
    a01 = wmma_bf16(fa0, fb1, a01);
    a10 = wmma_bf16(fa1, fb0, a10);
    a11 = wmma_bf16(fa1, fb1, a11);
  }

  #pragma unroll
  for (int r = 0; r < 8; ++r) {
    int d0 = half * 8 + r;
    int d1 = 16 + half * 8 + r;
    atomicAdd(&S[(size_t)(b * CKCH + d0) * CCH + c0 + col],      a00[r]);
    atomicAdd(&S[(size_t)(b * CKCH + d0) * CCH + c0 + 16 + col], a01[r]);
    atomicAdd(&S[(size_t)(b * CKCH + d1) * CCH + c0 + col],      a10[r]);
    atomicAdd(&S[(size_t)(b * CKCH + d1) * CCH + c0 + 16 + col], a11[r]);
  }
}

// ---------------- pass 3: context[d,e] = (Wv[e,:].S[d,:])/Z[d] + bv[e]  (tiny)
__global__ __launch_bounds__(256) void k_ctx(
    const float* __restrict__ S, const float* __restrict__ Z,
    const float* __restrict__ Wv, const float* __restrict__ bv,
    __bf16* __restrict__ ctxT)
{
  int idx = blockIdx.x * blockDim.x + threadIdx.x;
  if (idx >= NB * CKCH * CCH) return;
  int b = idx / (CKCH * CCH);
  int rem = idx % (CKCH * CCH);
  int d = rem / CCH;
  int e = rem % CCH;
  const float* Srow = S  + (size_t)(b * CKCH + d) * CCH;
  const float* Wrow = Wv + (size_t)e * CCH;
  float acc = 0.0f;
  for (int c = 0; c < CCH; ++c) acc += Srow[c] * Wrow[c];
  float ctx = acc / Z[b * CKCH + d] + bv[e];
  ctxT[((size_t)b * CCH + e) * CKCH + d] = to_bf16(ctx);   // [b][e][32]
}

// ------------- pass 4: out[e,n] = gamma * (ctx^T . q_sm)[e,n] + x[e,n]
// K = 32 == one bf16 WMMA; B fragment (q_sm) loaded once per column tile.
__global__ __launch_bounds__(256) void k_att(
    const float* __restrict__ x, const __bf16* __restrict__ ctxT,
    const __bf16* __restrict__ qsmT, const float* __restrict__ gamma,
    float* __restrict__ out)
{
  const int lane = threadIdx.x & 31;
  const int wave = threadIdx.x >> 5;
  const int half = lane >> 4;
  const int col  = lane & 15;
  const int b    = blockIdx.y;
  const int n0   = (blockIdx.x * 8 + wave) * 16;
  const float g  = gamma[0];
  const float* xb = x   + (size_t)b * CCH * NSP;
  float*       ob = out + (size_t)b * CCH * NSP;

  v16bf bf;
  const __bf16* qcol = qsmT + ((size_t)b * NSP + n0 + col) * CKCH;
  #pragma unroll
  for (int e = 0; e < 16; ++e) bf[e] = qcol[k_of(e, half)];

  for (int mt = 0; mt < 16; ++mt) {
    v16bf af;
    const __bf16* crow = ctxT + ((size_t)b * CCH + mt * 16 + col) * CKCH;
    #pragma unroll
    for (int e = 0; e < 16; ++e) af[e] = crow[k_of(e, half)];
    v8f acc = {};
    acc = wmma_bf16(af, bf, acc);
    #pragma unroll
    for (int r = 0; r < 8; ++r) {
      int ech = mt * 16 + half * 8 + r;
      size_t o = (size_t)ech * NSP + n0 + col;
      ob[o] = g * acc[r] + xb[o];
    }
  }
}

// -----------------------------------------------------------------------------
extern "C" void kernel_launch(void* const* d_in, const int* in_sizes, int n_in,
                              void* d_out, int out_size, void* d_ws, size_t ws_size,
                              hipStream_t stream) {
  const float* x     = (const float*)d_in[0];
  const float* Wq    = (const float*)d_in[1];
  const float* bq    = (const float*)d_in[2];
  const float* Wk    = (const float*)d_in[3];
  const float* bk    = (const float*)d_in[4];
  const float* Wv    = (const float*)d_in[5];
  const float* bv    = (const float*)d_in[6];
  const float* gamma = (const float*)d_in[7];
  float* out = (float*)d_out;

  // scratch layout (~24.2 MiB total)
  char* ws = (char*)d_ws;
  float*  klogit = (float*)ws;                                   // 16 MiB
  __bf16* qsmT   = (__bf16*)(ws + (size_t)NB * CKCH * NSP * 4);  // 8 MiB
  float*  S      = (float*)((char*)qsmT + (size_t)NB * NSP * CKCH * 2);
  float*  Z      = S + NB * CKCH * CCH;
  float*  kmax   = Z + NB * CKCH;
  __bf16* ctxT   = (__bf16*)(kmax + NB * CKCH);                  // 32 KiB
  __bf16* Wkbf   = ctxT + (size_t)NB * CCH * CKCH;               // 16 KiB
  __bf16* Wqbf   = Wkbf + (size_t)CKCH * CCH;

  k_init<<<dim3((NB * CKCH * CCH + 255) / 256), 256, 0, stream>>>(
      S, Z, kmax, Wk, Wq, Wkbf, Wqbf);
  k_qk  <<<dim3(NSP / 128, NB),   256, 0, stream>>>(x, Wqbf, bq, Wkbf, bk,
                                                    klogit, qsmT, kmax);
  k_sz  <<<dim3(NSP / CHUNK, NB), 256, 0, stream>>>(x, klogit, kmax, S, Z);
  k_ctx <<<dim3((NB * CKCH * CCH + 255) / 256), 256, 0, stream>>>(S, Z, Wv, bv, ctxT);
  k_att <<<dim3(NSP / 128, NB),   256, 0, stream>>>(x, ctxT, qsmT, gamma, out);
}
